// PointNet2SemSegMSG_66666482368660
// MI455X (gfx1250) — matrix operand order repaired and checked by reading, not compiled, not run on hardware
//
#include <hip/hip_runtime.h>
#include <hip/hip_bf16.h>

typedef __attribute__((ext_vector_type(16))) _Float16 v16h;
typedef __attribute__((ext_vector_type(8)))  float    v8f;

#define EPS_BN     1e-5f
#define EPS_INTERP 1e-8f

// ---------------------------------------------------------------------------
// WMMA GEMM: Y[M,Nc] = X[M,C] * W[Nc,C]^T   (f32 in/out, f16 MACs)
// Preconditions (guaranteed by host-side padding): M%32==0, C%32==0, Nc%32==0.
// One wave32 per 32x16 output tile (2 WMMA accumulators share one B fragment).
// All 12 tile loads issued before any conversion (single load clause);
// epilogue restaged through LDS for b128 stores.
// ---------------------------------------------------------------------------
__global__ void k_gemm_wmma(const float* __restrict__ X, const float* __restrict__ W,
                            float* __restrict__ Y, int M, int C, int Nc) {
  __shared__ alignas(16) _Float16 As[32][32];  // As[m][k]
  __shared__ alignas(16) _Float16 Bs[16][32];  // Bs[n][k] = W[n0+n, c0+k]
  __shared__ alignas(16) float    Ds[32][16];  // output restage
  const int lane = threadIdx.x;                // 0..31
  const int m0 = blockIdx.x * 32;
  const int n0 = blockIdx.y * 16;
  v8f acc0 = {}, acc1 = {};

  for (int c0 = 0; c0 < C; c0 += 32) {
    // ---- issue ALL tile loads first (A: 8 quads/lane, B: 4 quads/lane) ----
    float4 fa[8];
#pragma unroll
    for (int i = 0; i < 8; ++i) {
      int q   = i * 32 + lane;      // 0..255 float4 quads
      int row = q >> 3;             // 0..31
      int c4  = (q & 7) * 4;        // 0,4,...,28
      fa[i] = *(const float4*)(X + (size_t)(m0 + row) * C + c0 + c4);
    }
    float4 fb[4];
#pragma unroll
    for (int i = 0; i < 4; ++i) {
      int q   = i * 32 + lane;      // 0..127
      int row = q >> 3;             // 0..15
      int c4  = (q & 7) * 4;
      fb[i] = *(const float4*)(W + (size_t)(n0 + row) * C + c0 + c4);
    }
    // ---- convert + LDS store ----
#pragma unroll
    for (int i = 0; i < 8; ++i) {
      int q = i * 32 + lane; int row = q >> 3; int c4 = (q & 7) * 4;
      union { _Float16 h[4]; uint2 u; } pk;
      pk.h[0] = (_Float16)fa[i].x; pk.h[1] = (_Float16)fa[i].y;
      pk.h[2] = (_Float16)fa[i].z; pk.h[3] = (_Float16)fa[i].w;
      *(uint2*)&As[row][c4] = pk.u;
    }
#pragma unroll
    for (int i = 0; i < 4; ++i) {
      int q = i * 32 + lane; int row = q >> 3; int c4 = (q & 7) * 4;
      union { _Float16 h[4]; uint2 u; } pk;
      pk.h[0] = (_Float16)fb[i].x; pk.h[1] = (_Float16)fb[i].y;
      pk.h[2] = (_Float16)fb[i].z; pk.h[3] = (_Float16)fb[i].w;
      *(uint2*)&Bs[row][c4] = pk.u;
    }
    __syncthreads();

    // ---- fragments (contiguous b128 LDS reads) ----
    // A 16x32 f16 layout: lanes 0-15 hold M=r with K 0..7 / 16..23,
    //                     lanes 16-31 hold K 8..15 / 24..31.
    // B: same K layout as A (lanes 0-15 K 0..7 / 16..23, lanes 16-31 K 8..15 / 24..31), N = lane&15.
    const int r   = lane & 15;
    const int kb  = (lane >> 4) * 8;    // 0 or 8
    const int bkb = (lane >> 4) * 8;    // 0 or 8
    v16h a0, a1, b;
    {
      union { int4 q[2]; v16h v; } u;
      u.q[0] = *(const int4*)&As[r][kb];
      u.q[1] = *(const int4*)&As[r][16 + kb];
      a0 = u.v;
      u.q[0] = *(const int4*)&As[16 + r][kb];
      u.q[1] = *(const int4*)&As[16 + r][16 + kb];
      a1 = u.v;
      u.q[0] = *(const int4*)&Bs[r][bkb];
      u.q[1] = *(const int4*)&Bs[r][bkb + 16];
      b = u.v;
    }
    acc0 = __builtin_amdgcn_wmma_f32_16x16x32_f16(false, a0, false, b,
                                                  (short)0, acc0, false, false);
    acc1 = __builtin_amdgcn_wmma_f32_16x16x32_f16(false, a1, false, b,
                                                  (short)0, acc1, false, false);
    __syncthreads();
  }

  // ---- epilogue: restage through LDS, then b128 row stores ----
  // D layout: lane<16 -> N=lane, local rows base+v; lane>=16 -> rows base+8+v
  const int r  = lane & 15;
  const int rb = (lane >> 4) * 8;
#pragma unroll
  for (int v = 0; v < 8; ++v) {
    Ds[rb + v][r]      = acc0[v];
    Ds[16 + rb + v][r] = acc1[v];
  }
  __syncthreads();
  {
    float4* dst = (float4*)(Y + (size_t)(m0 + lane) * Nc + n0);
    const float4* src = (const float4*)&Ds[lane][0];
#pragma unroll
    for (int v = 0; v < 4; ++v) dst[v] = src[v];
  }
}

// ---------------------------------------------------------------------------
// zero-padded parameter staging
// ---------------------------------------------------------------------------
__global__ void k_pad_w(const float* __restrict__ W, float* __restrict__ Wp,
                        int Nc, int C, int Ncp, int Cp) {
  int g = blockIdx.x * blockDim.x + threadIdx.x;
  if (g >= Ncp * Cp) return;
  int c = g % Cp; int n = g / Cp;
  Wp[g] = (n < Nc && c < C) ? W[(size_t)n * C + c] : 0.f;
}

__global__ void k_pad_vec(const float* __restrict__ v, float* __restrict__ vp,
                          int Nv, int Np) {
  int g = blockIdx.x * blockDim.x + threadIdx.x;
  if (g >= Np) return;
  vp[g] = (g < Nv) ? v[g] : 0.f;
}

// ---------------------------------------------------------------------------
// BatchNorm (train-mode, per-channel over all rows) + ReLU
// ---------------------------------------------------------------------------
__global__ void k_bn_stats(const float* __restrict__ Y, float* __restrict__ mv,
                           int M, int C) {
  const int c = blockIdx.x;
  __shared__ float ss[256], ss2[256];
  float s = 0.f, s2 = 0.f;
  for (int i = threadIdx.x; i < M; i += 256) {
    float v = Y[(size_t)i * C + c];
    s += v; s2 += v * v;
  }
  ss[threadIdx.x] = s; ss2[threadIdx.x] = s2;
  __syncthreads();
  for (int off = 128; off > 0; off >>= 1) {
    if (threadIdx.x < off) {
      ss[threadIdx.x]  += ss[threadIdx.x + off];
      ss2[threadIdx.x] += ss2[threadIdx.x + off];
    }
    __syncthreads();
  }
  if (threadIdx.x == 0) {
    float mean = ss[0] / (float)M;
    mv[2 * c]     = mean;
    mv[2 * c + 1] = ss2[0] / (float)M - mean * mean;
  }
}

__global__ void k_bn_relu(float* __restrict__ Y, const float* __restrict__ mv,
                          const float* __restrict__ gamma, const float* __restrict__ beta,
                          unsigned long long tot, int C) {
  size_t g = (size_t)blockIdx.x * blockDim.x + threadIdx.x;
  if (g >= tot) return;
  int c = (int)(g % (size_t)C);
  float mean = mv[2 * c], var = mv[2 * c + 1];
  float v = (Y[g] - mean) * rsqrtf(var + EPS_BN) * gamma[c] + beta[c];
  Y[g] = fmaxf(v, 0.f);
}

// ---------------------------------------------------------------------------
// layout / gather kernels
// ---------------------------------------------------------------------------
__global__ void k_chlast(const float* __restrict__ in, float* __restrict__ out,
                         int B, int C, int N) {  // (B,C,N) -> (B,N,C)
  int g = blockIdx.x * blockDim.x + threadIdx.x;
  if (g >= B * C * N) return;
  int n = g % N; int c = (g / N) % C; int b = g / (N * C);
  out[((size_t)b * N + n) * C + c] = in[g];
}

__global__ void k_chfirst(const float* __restrict__ in, float* __restrict__ out,
                          int B, int N, int C, int strideC) { // (B,N,strideC) -> (B,C,N)
  int g = blockIdx.x * blockDim.x + threadIdx.x;
  if (g >= B * N * C) return;
  int c = g % C; int n = (g / C) % N; int b = g / (C * N);
  out[((size_t)b * C + c) * N + n] = in[((size_t)b * N + n) * strideC + c];
}

__global__ void k_gather3(const float* __restrict__ pts, const int* __restrict__ idx,
                          float* __restrict__ out, int B, int N, int S) {
  int g = blockIdx.x * blockDim.x + threadIdx.x;
  if (g >= B * S * 3) return;
  int c = g % 3; int s = (g / 3) % S; int b = g / (3 * S);
  int id = idx[b * S + s];
  out[g] = pts[((size_t)b * N + id) * 3 + c];
}

// ---------------------------------------------------------------------------
// farthest point sampling: one 256-thread block per batch
// ---------------------------------------------------------------------------
__global__ void k_fps(const float* __restrict__ pts, int* __restrict__ out_idx,
                      int N, int S) {
  const int b = blockIdx.x;
  const float* p = pts + (size_t)b * N * 3;
  __shared__ float sval[256];
  __shared__ int   sidx[256];
  __shared__ float cx, cy, cz;
  __shared__ int   curFar;
  float mind[32];
  const int t = threadIdx.x;
  const int per = (N + 255) / 256;
  for (int j = 0; j < per; ++j) mind[j] = 1e10f;
  if (t == 0) curFar = 0;
  __syncthreads();
  for (int s = 0; s < S; ++s) {
    if (t == 0) {
      out_idx[b * S + s] = curFar;
      cx = p[(size_t)curFar * 3];
      cy = p[(size_t)curFar * 3 + 1];
      cz = p[(size_t)curFar * 3 + 2];
    }
    __syncthreads();
    float best = -1.f; int bi = 0;
    for (int j = 0; j < per; ++j) {
      int i = t + j * 256;
      if (i < N) {
        float dx = p[(size_t)i * 3]     - cx;
        float dy = p[(size_t)i * 3 + 1] - cy;
        float dz = p[(size_t)i * 3 + 2] - cz;
        float d = dx * dx + dy * dy + dz * dz;
        mind[j] = fminf(mind[j], d);
        if (mind[j] > best) { best = mind[j]; bi = i; }
      }
    }
    sval[t] = best; sidx[t] = bi;
    __syncthreads();
    for (int off = 128; off > 0; off >>= 1) {
      if (t < off && sval[t + off] > sval[t]) {
        sval[t] = sval[t + off]; sidx[t] = sidx[t + off];
      }
      __syncthreads();
    }
    if (t == 0) curFar = sidx[0];
    __syncthreads();
  }
}

// ---------------------------------------------------------------------------
// ball query: first K indices (original order) within radius, pad with first hit
// ---------------------------------------------------------------------------
__global__ void k_ballquery(const float* __restrict__ pts, const float* __restrict__ q,
                            int* __restrict__ idx, int B, int N, int S, int K, float r2) {
  int g = blockIdx.x * blockDim.x + threadIdx.x;
  if (g >= B * S) return;
  int b = g / S;
  const float* p = pts + (size_t)b * N * 3;
  float qx = q[(size_t)g * 3], qy = q[(size_t)g * 3 + 1], qz = q[(size_t)g * 3 + 2];
  int* o = idx + (size_t)g * K;
  int cnt = 0, first = 0; bool has = false;
  for (int i = 0; i < N && cnt < K; ++i) {
    float dx = p[(size_t)i * 3] - qx;
    float dy = p[(size_t)i * 3 + 1] - qy;
    float dz = p[(size_t)i * 3 + 2] - qz;
    if (dx * dx + dy * dy + dz * dz < r2) {
      if (!has) { first = i; has = true; }
      o[cnt++] = i;
    }
  }
  for (; cnt < K; ++cnt) o[cnt] = first;
}

// ---------------------------------------------------------------------------
// group: X[row, 0:3] = xyz[idx]-new_xyz ; X[row, 3:3+Cf] = feats[idx]
// X row stride = Cinp (padded); pad columns left as-is (multiplied by zero W)
// ---------------------------------------------------------------------------
__global__ void k_group(const float* __restrict__ pts, const float* __restrict__ newxyz,
                        const float* __restrict__ feats, const int* __restrict__ idx,
                        float* __restrict__ X, int B, int N, int S, int K, int Cf, int Cinp) {
  int g = blockIdx.x * blockDim.x + threadIdx.x;
  if (g >= B * S * K) return;
  int s = (g / K) % S; int b = g / (K * S);
  int id = idx[g];
  const float* p = pts    + ((size_t)b * N + id) * 3;
  const float* q = newxyz + ((size_t)b * S + s) * 3;
  float* x = X + (size_t)g * Cinp;
  x[0] = p[0] - q[0]; x[1] = p[1] - q[1]; x[2] = p[2] - q[2];
  if (Cf > 0) {
    const float* f = feats + ((size_t)b * N + id) * Cf;
    for (int c = 0; c < Cf; ++c) x[3 + c] = f[c];
  }
}

// ---------------------------------------------------------------------------
// max-pool over K group members (input row stride strideC), write into concat
// ---------------------------------------------------------------------------
__global__ void k_maxpool(const float* __restrict__ Y, float* __restrict__ out,
                          int BS, int K, int C, int strideC, int outC, int coff) {
  int g = blockIdx.x * blockDim.x + threadIdx.x;
  if (g >= BS * C) return;
  int c = g % C; int s = g / C;
  float m = -3.4e38f;
  const float* p = Y + (size_t)s * K * strideC + c;
  for (int k = 0; k < K; ++k) m = fmaxf(m, p[(size_t)k * strideC]);
  out[(size_t)s * outC + coff + c] = m;
}

// ---------------------------------------------------------------------------
// 3-NN (squared distances, like the reference) + inverse-distance weights
// ---------------------------------------------------------------------------
__global__ void k_three_nn(const float* __restrict__ uq, const float* __restrict__ kn,
                           int* __restrict__ nidx, float* __restrict__ nw,
                           int B, int n, int m) {
  int g = blockIdx.x * blockDim.x + threadIdx.x;
  if (g >= B * n) return;
  int b = g / n;
  float ux = uq[(size_t)g * 3], uy = uq[(size_t)g * 3 + 1], uz = uq[(size_t)g * 3 + 2];
  const float* kp = kn + (size_t)b * m * 3;
  float d0 = 3.4e38f, d1 = 3.4e38f, d2 = 3.4e38f;
  int   i0 = 0, i1 = 0, i2 = 0;
  for (int j = 0; j < m; ++j) {
    float dx = ux - kp[(size_t)j * 3];
    float dy = uy - kp[(size_t)j * 3 + 1];
    float dz = uz - kp[(size_t)j * 3 + 2];
    float d = dx * dx + dy * dy + dz * dz;
    if (d < d0)      { d2 = d1; i2 = i1; d1 = d0; i1 = i0; d0 = d; i0 = j; }
    else if (d < d1) { d2 = d1; i2 = i1; d1 = d;  i1 = j; }
    else if (d < d2) { d2 = d;  i2 = j; }
  }
  float r0 = 1.f / (d0 + EPS_INTERP);
  float r1 = 1.f / (d1 + EPS_INTERP);
  float r2 = 1.f / (d2 + EPS_INTERP);
  float rs = r0 + r1 + r2;
  nidx[(size_t)g * 3] = i0; nidx[(size_t)g * 3 + 1] = i1; nidx[(size_t)g * 3 + 2] = i2;
  nw[(size_t)g * 3] = r0 / rs; nw[(size_t)g * 3 + 1] = r1 / rs; nw[(size_t)g * 3 + 2] = r2 / rs;
}

// known features have row stride kfStride; write into X (row stride outC) at coff
__global__ void k_interp(const float* __restrict__ kf, const int* __restrict__ nidx,
                         const float* __restrict__ nw, float* __restrict__ X,
                         int B, int n, int m, int C, int kfStride, int outC, int coff) {
  size_t g = (size_t)blockIdx.x * blockDim.x + threadIdx.x;
  size_t tot = (size_t)B * n * C;
  if (g >= tot) return;
  int c = (int)(g % (size_t)C);
  size_t r = g / (size_t)C;
  int b = (int)(r / (size_t)n);
  const int*   id = nidx + r * 3;
  const float* w  = nw   + r * 3;
  const float* f  = kf + (size_t)b * m * kfStride;
  float v = w[0] * f[(size_t)id[0] * kfStride + c]
          + w[1] * f[(size_t)id[1] * kfStride + c]
          + w[2] * f[(size_t)id[2] * kfStride + c];
  X[r * outC + coff + c] = v;
}

__global__ void k_copy_cols(const float* __restrict__ src, float* __restrict__ dst,
                            int rows, int C, int srcStride, int outC, int coff) {
  size_t g = (size_t)blockIdx.x * blockDim.x + threadIdx.x;
  size_t tot = (size_t)rows * C;
  if (g >= tot) return;
  int c = (int)(g % (size_t)C);
  size_t r = g / (size_t)C;
  dst[r * outC + coff + c] = src[r * srcStride + c];
}

// ---------------------------------------------------------------------------
// host orchestration
// ---------------------------------------------------------------------------
extern "C" void kernel_launch(void* const* d_in, const int* in_sizes, int n_in,
                              void* d_out, int out_size, void* d_ws, size_t ws_size,
                              hipStream_t stream) {
  (void)in_sizes; (void)n_in; (void)out_size; (void)ws_size;
  const int B = 4, N = 8192;
  const float* xyz = (const float*)d_in[0];

  struct Layer { const float* W; const float* g; const float* b; int cin, cout; };
  int p = 1;
  auto take = [&](int cin, int cout) {
    Layer L{(const float*)d_in[p], (const float*)d_in[p + 1], (const float*)d_in[p + 2],
            cin, cout};
    p += 3;
    return L;
  };

  const int SA1_CH[3][4] = {{3, 32, 32, 64}, {3, 64, 64, 128}, {3, 64, 96, 128}};
  const int SA2_CH[2][4] = {{323, 128, 128, 256}, {323, 128, 196, 256}};
  const int FP1_CH[3] = {832, 256, 256};
  const int FP2_CH[4] = {256, 256, 128, 2};

  Layer sa1[3][3], sa2[2][3], fp1[2], fp2[3];
  for (int br = 0; br < 3; ++br)
    for (int l = 0; l < 3; ++l) sa1[br][l] = take(SA1_CH[br][l], SA1_CH[br][l + 1]);
  for (int br = 0; br < 2; ++br)
    for (int l = 0; l < 3; ++l) sa2[br][l] = take(SA2_CH[br][l], SA2_CH[br][l + 1]);
  for (int l = 0; l < 2; ++l) fp1[l] = take(FP1_CH[l], FP1_CH[l + 1]);
  for (int l = 0; l < 3; ++l) fp2[l] = take(FP2_CH[l], FP2_CH[l + 1]);

  auto a32  = [](int x) { return (x + 31) & ~31; };
  auto cdiv = [](int a, int b) { return (a + b - 1) / b; };

  // ---- workspace arena ----
  char* base = (char*)d_ws;
  size_t off = 0;
  auto alloc_f = [&](size_t n) -> float* {
    off = (off + 255) & ~(size_t)255;
    float* r = (float*)(base + off);
    off += n * sizeof(float);
    return r;
  };
  auto alloc_i = [&](size_t n) -> int* {
    off = (off + 255) & ~(size_t)255;
    int* r = (int*)(base + off);
    off += n * sizeof(int);
    return r;
  };

  float* pts    = alloc_f((size_t)B * N * 3);
  float* l1_xyz = alloc_f((size_t)B * 1024 * 3);
  float* l1_f   = alloc_f((size_t)B * 1024 * 320);
  float* l2_xyz = alloc_f((size_t)B * 256 * 3);
  float* l2_f   = alloc_f((size_t)B * 256 * 512);
  float* l1_f2  = alloc_f((size_t)B * 1024 * 256);
  int*   fpsidx = alloc_i((size_t)B * 1024);
  int*   bqidx  = alloc_i((size_t)B * 1024 * 128);
  int*   nnidx  = alloc_i((size_t)B * 8192 * 3);
  float* nnw    = alloc_f((size_t)B * 8192 * 3);
  float* mv     = alloc_f(1024);
  float* wpad   = alloc_f((size_t)256 * 832);            // max padded weight
  float* gpad   = alloc_f(512);
  float* bpad   = alloc_f(512);
  float* tmpA   = alloc_f((size_t)B * 1024 * 128 * 96);  // ping (max 50.3M f)
  float* tmpB   = alloc_f((size_t)B * 1024 * 128 * 128); // pong (max 67.1M f)

  auto run_mlp = [&](int M, const Layer* Ls, int nl) -> float* {
    float* cur = tmpA; float* nxt = tmpB;
    for (int l = 0; l < nl; ++l) {
      const int cinp  = a32(Ls[l].cin);
      const int coutp = a32(Ls[l].cout);
      { int tot = coutp * cinp;
        k_pad_w<<<cdiv(tot, 256), 256, 0, stream>>>(Ls[l].W, wpad, Ls[l].cout, Ls[l].cin,
                                                    coutp, cinp); }
      k_pad_vec<<<cdiv(coutp, 256), 256, 0, stream>>>(Ls[l].g, gpad, Ls[l].cout, coutp);
      k_pad_vec<<<cdiv(coutp, 256), 256, 0, stream>>>(Ls[l].b, bpad, Ls[l].cout, coutp);
      dim3 grid(M / 32, coutp / 16);
      k_gemm_wmma<<<grid, 32, 0, stream>>>(cur, wpad, nxt, M, cinp, coutp);
      k_bn_stats<<<coutp, 256, 0, stream>>>(nxt, mv, M, coutp);
      size_t tot = (size_t)M * coutp;
      k_bn_relu<<<(unsigned)((tot + 255) / 256), 256, 0, stream>>>(
          nxt, mv, gpad, bpad, (unsigned long long)tot, coutp);
      float* t = cur; cur = nxt; nxt = t;
    }
    return cur;
  };

  // ---- transpose input (B,3,N) -> (B,N,3) ----
  { int tot = B * 3 * N; k_chlast<<<cdiv(tot, 256), 256, 0, stream>>>(xyz, pts, B, 3, N); }

  // ================= SA1 (MSG) =================
  k_fps<<<B, 256, 0, stream>>>(pts, fpsidx, N, 1024);
  { int tot = B * 1024 * 3;
    k_gather3<<<cdiv(tot, 256), 256, 0, stream>>>(pts, fpsidx, l1_xyz, B, N, 1024); }
  const float RAD1[3] = {0.1f, 0.2f, 0.4f};
  const int   NS1[3]  = {16, 32, 128};
  int coff = 0;
  for (int br = 0; br < 3; ++br) {
    const int S = 1024, K = NS1[br];
    { int tot = B * S;
      k_ballquery<<<cdiv(tot, 256), 256, 0, stream>>>(pts, l1_xyz, bqidx, B, N, S, K,
                                                      RAD1[br] * RAD1[br]); }
    const int M = B * S * K;
    k_group<<<cdiv(M, 256), 256, 0, stream>>>(pts, l1_xyz, (const float*)nullptr, bqidx,
                                              tmpA, B, N, S, K, 0, a32(3));
    float* Yf = run_mlp(M, sa1[br], 3);
    const int Cl = SA1_CH[br][3];
    { int tot = B * S * Cl;
      k_maxpool<<<cdiv(tot, 256), 256, 0, stream>>>(Yf, l1_f, B * S, K, Cl, a32(Cl),
                                                    320, coff); }
    coff += Cl;
  }

  // ================= SA2 (MSG) =================
  k_fps<<<B, 256, 0, stream>>>(l1_xyz, fpsidx, 1024, 256);
  { int tot = B * 256 * 3;
    k_gather3<<<cdiv(tot, 256), 256, 0, stream>>>(l1_xyz, fpsidx, l2_xyz, B, 1024, 256); }
  const float RAD2[2] = {0.4f, 0.8f};
  const int   NS2[2]  = {32, 64};
  coff = 0;
  for (int br = 0; br < 2; ++br) {
    const int S = 256, K = NS2[br], Np = 1024;
    { int tot = B * S;
      k_ballquery<<<cdiv(tot, 256), 256, 0, stream>>>(l1_xyz, l2_xyz, bqidx, B, Np, S, K,
                                                      RAD2[br] * RAD2[br]); }
    const int M = B * S * K;
    k_group<<<cdiv(M, 256), 256, 0, stream>>>(l1_xyz, l2_xyz, l1_f, bqidx, tmpA,
                                              B, Np, S, K, 320, a32(323));
    float* Yf = run_mlp(M, sa2[br], 3);
    const int Cl = 256;
    { int tot = B * S * Cl;
      k_maxpool<<<cdiv(tot, 256), 256, 0, stream>>>(Yf, l2_f, B * S, K, Cl, a32(Cl),
                                                    512, coff); }
    coff += Cl;
  }

  // ================= FP1: l2 -> l1 =================
  { int tot = B * 1024;
    k_three_nn<<<cdiv(tot, 256), 256, 0, stream>>>(l1_xyz, l2_xyz, nnidx, nnw,
                                                   B, 1024, 256); }
  { size_t tot = (size_t)B * 1024 * 512;
    k_interp<<<(unsigned)((tot + 255) / 256), 256, 0, stream>>>(
        l2_f, nnidx, nnw, tmpA, B, 1024, 256, 512, 512, 832, 0); }
  { size_t tot = (size_t)B * 1024 * 320;
    k_copy_cols<<<(unsigned)((tot + 255) / 256), 256, 0, stream>>>(
        l1_f, tmpA, B * 1024, 320, 320, 832, 512); }
  {
    float* Yf = run_mlp(B * 1024, fp1, 2);  // ends back in tmpA (stride 256)
    size_t tot = (size_t)B * 1024 * 256;
    k_copy_cols<<<(unsigned)((tot + 255) / 256), 256, 0, stream>>>(
        Yf, l1_f2, B * 1024, 256, 256, 256, 0);
  }

  // ================= FP2: l1 -> l0 =================
  { int tot = B * N;
    k_three_nn<<<cdiv(tot, 256), 256, 0, stream>>>(pts, l1_xyz, nnidx, nnw, B, N, 1024); }
  { size_t tot = (size_t)B * N * 256;
    k_interp<<<(unsigned)((tot + 255) / 256), 256, 0, stream>>>(
        l1_f2, nnidx, nnw, tmpA, B, N, 1024, 256, 256, 256, 0); }
  float* Yo = run_mlp(B * N, fp2, 3);  // 3 layers -> result in tmpB (stride a32(2)=32)
  { int tot = B * N * 2;
    k_chfirst<<<cdiv(tot, 256), 256, 0, stream>>>(Yo, (float*)d_out, B, N, 2, 32); }
}
